// LateInteraction_34720515621483
// MI455X (gfx1250) — matrix operand order repaired and checked by reading, not compiled, not run on hardware
//
#include <hip/hip_runtime.h>
#include <hip/hip_bf16.h>

typedef _Float16 half_t;
typedef _Float16 v8h  __attribute__((ext_vector_type(8)));
typedef _Float16 v16h __attribute__((ext_vector_type(16)));
typedef float    v8f  __attribute__((ext_vector_type(8)));

#define BB 32
#define SS 256
#define HH 128
#define SIM_STRIDE 260   // 256 + 4 pad to break LDS bank patterns

// ---------------------------------------------------------------------------
// Pass 1: L2-normalize each token embedding (H=128) and convert to f16.
// One block per token, 128 threads (4 waves of 32).
// ---------------------------------------------------------------------------
__global__ __launch_bounds__(128)
void li_normalize_f16_kernel(const float* __restrict__ in,
                             half_t* __restrict__ out) {
    const int tok = blockIdx.x;
    const int t   = threadIdx.x;          // 0..127
    const float x = in[(size_t)tok * HH + t];
    float ss = x * x;
    // wave32 butterfly sum
    #pragma unroll
    for (int off = 16; off >= 1; off >>= 1)
        ss += __shfl_xor(ss, off, 32);
    __shared__ float wss[4];
    const int wid = t >> 5;
    if ((t & 31) == 0) wss[wid] = ss;
    __syncthreads();
    const float tot  = wss[0] + wss[1] + wss[2] + wss[3];
    const float rinv = 1.0f / fmaxf(sqrtf(tot), 1e-12f);
    out[(size_t)tok * HH + t] = (half_t)(x * rinv);
}

// ---------------------------------------------------------------------------
// Pass 2: per (i,j) pair, compute sim = Qi * Kj^T via WMMA (K=128),
// fused distance-decay + mask + softmax over t, and the softmax-weighted
// mean of raw sims (algebraic collapse of the p@K / q·agg GEMMs).
// 512 threads = 16 waves; wave w owns t-tile w in the WMMA phase and
// s-row w in the softmax phase.
// ---------------------------------------------------------------------------
__global__ __launch_bounds__(512)
void li_scores_kernel(const half_t* __restrict__ qh,
                      const half_t* __restrict__ kh,
                      const float* __restrict__ q_mask,
                      const float* __restrict__ k_mask,
                      const float* __restrict__ alpha_raw,
                      const float* __restrict__ logit_scale,
                      float* __restrict__ out) {
    __shared__ float sim_lds[16 * SIM_STRIDE];
    __shared__ float qm_sh[SS];
    __shared__ float km_sh[SS];
    __shared__ float red[16];

    const int i    = blockIdx.y;
    const int j    = blockIdx.x;
    const int tid  = threadIdx.x;
    const int w    = tid >> 5;        // wave id 0..15
    const int lane = tid & 31;
    const int r    = lane & 15;
    const int hi   = lane >> 4;

    // stage masks for this (i,j)
    if (tid < SS)            qm_sh[tid]      = q_mask[(size_t)i * SS + tid];
    else if (tid < 2 * SS)   km_sh[tid - SS] = k_mask[(size_t)j * SS + (tid - SS)];
    __syncthreads();

    const float araw  = alpha_raw[0];
    const float alpha = (araw > 20.0f) ? araw : log1pf(__expf(araw)); // softplus
    const float scale = __expf(logit_scale[0]);

    const half_t* qbase = qh + (size_t)i * SS * HH;
    const half_t* kbase = kh + (size_t)j * SS * HH;

    float acc_dots = 0.0f;

    for (int st = 0; st < SS / 16; ++st) {
        // ---- WMMA phase: 16x16 tile per wave, K=128 in 4 chunks of 32 ----
        v8f acc = {};
        const int sRow = st * 16 + r;   // A-matrix row (query token)
        const int tRow = w  * 16 + r;   // B-matrix col (key token)
        #pragma unroll
        for (int kc = 0; kc < 4; ++kc) {
            const int k0 = kc * 32;
            // A (16x32 f16): lane hi-half interleave per ISA layout
            const half_t* pa = qbase + (size_t)sRow * HH + k0 + 8 * hi;
            v8h a0 = *(const v8h*)(pa);
            v8h a1 = *(const v8h*)(pa + 16);
            // B (32x16 f16): lane hi-half interleave per ISA layout (same K order as A)
            const half_t* pb = kbase + (size_t)tRow * HH + k0 + 8 * hi;
            v8h b0 = *(const v8h*)(pb);
            v8h b1 = *(const v8h*)(pb + 16);
            v16h a, b;
            #pragma unroll
            for (int e = 0; e < 8; ++e) {
                a[e] = a0[e]; a[e + 8] = a1[e];
                b[e] = b0[e]; b[e + 8] = b1[e];
            }
            acc = __builtin_amdgcn_wmma_f32_16x16x32_f16(
                      false, a, false, b, (short)0, acc, false, false);
        }
        // D layout: VGPR v -> row (v + 8*hi), col r
        #pragma unroll
        for (int v = 0; v < 8; ++v)
            sim_lds[(v + 8 * hi) * SIM_STRIDE + w * 16 + r] = acc[v];
        __syncthreads();

        // ---- softmax phase: wave w handles s-row w of this tile ----
        {
            const int   s  = st * 16 + w;
            const float qm = qm_sh[s];
            float simv[8], lg[8];
            float mx = -3.0e38f;
            #pragma unroll
            for (int c = 0; c < 8; ++c) {
                const int   t  = lane + 32 * c;          // conflict-free LDS read
                const float sv = sim_lds[w * SIM_STRIDE + t];
                simv[c] = sv;
                const bool  valid = (qm * km_sh[t]) > 0.0f;
                const float dist  = fabsf((float)(s - t));
                lg[c] = valid ? scale * __expf(-alpha * dist) * sv : -1.0e9f;
                mx = fmaxf(mx, lg[c]);
            }
            #pragma unroll
            for (int off = 16; off >= 1; off >>= 1)
                mx = fmaxf(mx, __shfl_xor(mx, off, 32));
            float sum = 0.0f, wsum = 0.0f;
            #pragma unroll
            for (int c = 0; c < 8; ++c) {
                const float p = __expf(lg[c] - mx);
                sum  += p;
                wsum += p * simv[c];
            }
            #pragma unroll
            for (int off = 16; off >= 1; off >>= 1) {
                sum  += __shfl_xor(sum,  off, 32);
                wsum += __shfl_xor(wsum, off, 32);
            }
            acc_dots += (wsum / sum) * qm;   // dots[i,j,s] * q_mask[i,s]
        }
        __syncthreads();
    }

    if (lane == 0) red[w] = acc_dots;
    __syncthreads();
    if (tid == 0) {
        float tot = 0.0f;
        #pragma unroll
        for (int v = 0; v < 16; ++v) tot += red[v];
        float qsum = 0.0f;
        for (int s = 0; s < SS; ++s) qsum += qm_sh[s];
        out[(size_t)i * BB + j] = tot / fmaxf(qsum, 1.0f);
    }
}

// ---------------------------------------------------------------------------
extern "C" void kernel_launch(void* const* d_in, const int* in_sizes, int n_in,
                              void* d_out, int out_size, void* d_ws, size_t ws_size,
                              hipStream_t stream) {
    const float* q_embs      = (const float*)d_in[0];
    const float* k_embs      = (const float*)d_in[1];
    const float* q_mask      = (const float*)d_in[2];
    const float* k_mask      = (const float*)d_in[3];
    const float* alpha_raw   = (const float*)d_in[4];
    const float* logit_scale = (const float*)d_in[5];
    float* out = (float*)d_out;

    half_t* qh = (half_t*)d_ws;                       // B*S*H f16 = 2 MB
    half_t* kh = qh + (size_t)BB * SS * HH;           // + 2 MB

    li_normalize_f16_kernel<<<BB * SS, 128, 0, stream>>>(q_embs, qh);
    li_normalize_f16_kernel<<<BB * SS, 128, 0, stream>>>(k_embs, kh);

    dim3 grid(BB, BB);
    li_scores_kernel<<<grid, 512, 0, stream>>>(qh, kh, q_mask, k_mask,
                                               alpha_raw, logit_scale, out);
}
